// NeRFRenderer_8538394984933
// MI455X (gfx1250) — matrix operand order repaired and checked by reading, not compiled, not run on hardware
//
#include <hip/hip_runtime.h>
#include <cmath>

typedef __attribute__((ext_vector_type(16))) _Float16 v16h;
typedef __attribute__((ext_vector_type(8)))  float    v8f;
typedef __attribute__((ext_vector_type(2)))  float    v2f;
typedef __attribute__((ext_vector_type(4)))  unsigned int u32x4;
typedef __attribute__((ext_vector_type(8)))  int      i32x8;
typedef __attribute__((ext_vector_type(4)))  int      i32x4;

#define NRAYS 8192
#define SC 128          // coarse steps
#define SU 128          // upsample (fine) steps
#define ST 256          // merged steps
#define HID 64
#define GEO 15
#define ZNEAR 0.2f
#define ZFAR  (0.2f * 81.0f)
#define BND 20.0f

__device__ __forceinline__ float clampb(float x) {
    return fminf(fmaxf(x, -BND), BND);
}

// ---------------------------------------------------------------------------
// Density MLP: x(3) -> relu(64) -> 16  (sigma = exp(o0), geo = o1..15)
// Layer1: exact f32 path, 4x v_wmma_f32_16x16x4_f32 (K=4 covers the 3 dims).
// Layer2: 2x v_wmma_f32_16x16x32_f16 over K=64; Wd2 staged to LDS by the TDM.
// Hidden activations repacked C-layout -> A-layout through LDS.
// ---------------------------------------------------------------------------
__global__ __launch_bounds__(128)
void density_mlp(const float* __restrict__ rays_o, const float* __restrict__ rays_d,
                 const float* __restrict__ zbuf,       // null -> uniform coarse z
                 const float* __restrict__ Wd1, const float* __restrict__ bd1,
                 const float* __restrict__ Wd2, const float* __restrict__ bd2,
                 float* __restrict__ sigma_out, _Float16* __restrict__ geo_out,
                 int steps, int n_tiles)
{
    __shared__ _Float16 lds_h[4][16 * HID];
    __shared__ float    lds_w2[HID * 16];     // TDM-staged Wd2 (64x16 f32)
    const int lane = threadIdx.x & 31;
    const int wid  = threadIdx.x >> 5;
    const int col  = lane & 15;
    const int hi   = (lane >= 16) ? 1 : 0;

    __builtin_prefetch(Wd1, 0, 0);

    // ---- TDM: DMA the 64x16 f32 Wd2 tile into LDS (one issue per workgroup) ----
    if (wid == 0) {
        unsigned long long ga = (unsigned long long)(uintptr_t)Wd2;
        unsigned int lb = (unsigned int)(uintptr_t)&lds_w2[0];  // flat addr low 32b = LDS offset
        u32x4 g0 = {};
        g0[0] = 1u;                                    // count = 1 valid descriptor
        g0[1] = lb;                                    // lds_addr
        g0[2] = (unsigned int)(ga & 0xffffffffu);      // global_addr[31:0]
        g0[3] = (unsigned int)((ga >> 32) & 0x1ffffffu) | (2u << 30);  // [56:32] | type=2
        i32x8 g1 = {};
        g1[0] = (int)(2u << 16);                       // workgroup_mask=0, data_size=4B
        g1[1] = (int)(16u << 16);                      // tensor_dim0 = 16 (low16)
        g1[2] = (int)(64u << 16);                      // dim0 hi=0 ; tensor_dim1 = 64 (low16)
        g1[3] = (int)(16u << 16);                      // dim1 hi=0 ; tile_dim0 = 16
        g1[4] = (int)64;                               // tile_dim1 = 64 ; tile_dim2 = 0
        g1[5] = (int)16;                               // tensor_dim0_stride = 16
        g1[6] = 0; g1[7] = 0;
        i32x4 z4 = {};
#if defined(__clang_major__) && (__clang_major__ >= 23)
        i32x8 z8 = {};
        __builtin_amdgcn_tensor_load_to_lds(g0, g1, z4, z4, z8, 0);
#else
        __builtin_amdgcn_tensor_load_to_lds(g0, g1, z4, z4, 0);
#endif
        __builtin_amdgcn_s_wait_tensorcnt(0);
    }
    __syncthreads();

    // ---- B fragments ----
    // Layer1 (f32 WMMA, 16x16x4): B is 4x16; VGPR r -> K = r + 2*hi, N = col.
    v2f b1[4];
#pragma unroll
    for (int t = 0; t < 4; t++) {
        v2f b;
        int K0 = 2 * hi;      // 0 or 2  (always < 3: valid)
        int K1 = 2 * hi + 1;  // 1 or 3  (3 is pad)
        b[0] = Wd1[K0 * HID + 16 * t + col];
        b[1] = (K1 < 3) ? Wd1[K1 * HID + 16 * t + col] : 0.f;
        b1[t] = b;
    }
    // Layer2 (f16 WMMA, 16x16x32): B from TDM-staged LDS copy of Wd2.
    v16h b2[2];
#pragma unroll
    for (int c = 0; c < 2; c++) {
        v16h b = {};
#pragma unroll
        for (int e = 0; e < 16; e++) {
            int K = 32 * c + hi * 8 + (e < 8 ? e : e + 8);
            b[e] = (_Float16)lds_w2[K * 16 + col];
        }
        b2[c] = b;
    }
    float bias1[4];
#pragma unroll
    for (int t = 0; t < 4; t++) bias1[t] = bd1[16 * t + col];
    const float bias2 = bd2[col];

    const float dzc = (ZFAR - ZNEAR) / (float)(SC - 1);

    int wglobal = blockIdx.x * 4 + wid;
    int wstride = gridDim.x * 4;
    for (int tile = wglobal; tile < n_tiles; tile += wstride) {
        // ---- Build A (16x4 f32): rows = samples; lane<16 gives K0,K1; lane>=16 K2,K3 ----
        const int row = col;
        long g  = (long)tile * 16 + row;
        int ray = (int)(g / steps);
        int s   = (int)(g % steps);
        float z = zbuf ? zbuf[g] : (ZNEAR + dzc * (float)s);
        float px = clampb(rays_o[ray * 3 + 0] + rays_d[ray * 3 + 0] * z);
        float py = clampb(rays_o[ray * 3 + 1] + rays_d[ray * 3 + 1] * z);
        float pz = clampb(rays_o[ray * 3 + 2] + rays_d[ray * 3 + 2] * z);
        v2f a1;
        a1[0] = hi ? pz : px;     // K2 : K0
        a1[1] = hi ? 0.f : py;    // K3(pad) : K1

        v8f cacc[4] = {{}, {}, {}, {}};
        cacc[0] = __builtin_amdgcn_wmma_f32_16x16x4_f32(false, a1, false, b1[0], (short)0, cacc[0], false, false);
        cacc[1] = __builtin_amdgcn_wmma_f32_16x16x4_f32(false, a1, false, b1[1], (short)0, cacc[1], false, false);
        cacc[2] = __builtin_amdgcn_wmma_f32_16x16x4_f32(false, a1, false, b1[2], (short)0, cacc[2], false, false);
        cacc[3] = __builtin_amdgcn_wmma_f32_16x16x4_f32(false, a1, false, b1[3], (short)0, cacc[3], false, false);

        // ---- bias + relu, pack hidden (16x64) to LDS in row-major f16 ----
        _Float16* L = lds_h[wid];
#pragma unroll
        for (int t = 0; t < 4; t++)
#pragma unroll
            for (int r = 0; r < 8; r++) {
                float v = cacc[t][r] + bias1[t];
                v = v > 0.f ? v : 0.f;
                int rowm = r + hi * 8;                // C/D: M = r + 8*(lane>=16)
                L[rowm * HID + 16 * t + col] = (_Float16)v;
            }
        asm volatile("s_wait_dscnt 0" ::: "memory");  // LDS stores visible before reads

        // ---- Layer2: A = hidden (16x64 f16), two K-chunks of 32 ----
        v16h a2[2];
#pragma unroll
        for (int c = 0; c < 2; c++) {
            v16h a = {};
            int kb = 32 * c + hi * 8;                 // A: K base per half-wave
#pragma unroll
            for (int e = 0; e < 16; e++) {
                int K = kb + (e < 8 ? e : e + 8);     // A: e<8 -> K, e>=8 -> K+8 skip
                a[e] = L[row * HID + K];
            }
            a2[c] = a;
        }
        v8f o = {};
        o = __builtin_amdgcn_wmma_f32_16x16x32_f16(false, a2[0], false, b2[0], (short)0, o, false, false);
        o = __builtin_amdgcn_wmma_f32_16x16x32_f16(false, a2[1], false, b2[1], (short)0, o, false, false);

#pragma unroll
        for (int r = 0; r < 8; r++) {
            int rowm = r + hi * 8;
            long g2 = (long)tile * 16 + rowm;
            float v = o[r] + bias2;
            if (col == 0) sigma_out[g2] = expf(v);
            else          geo_out[g2 * GEO + (col - 1)] = (_Float16)v;
        }
    }
}

// ---------------------------------------------------------------------------
// Per-ray: coarse compositing + deterministic inverse-CDF importance sampling.
// ---------------------------------------------------------------------------
__global__ __launch_bounds__(256)
void pdf_kernel(const float* __restrict__ sigma, float* __restrict__ newz)
{
    int ray = blockIdx.x * blockDim.x + threadIdx.x;
    if (ray >= NRAYS) return;
    const float dz = (ZFAR - ZNEAR) / (float)(SC - 1);
    const float sample_dist = (ZFAR - ZNEAR) / (float)SC;

    float w[SC];
    float T = 1.f;
    for (int i = 0; i < SC; i++) {
        float delta = (i < SC - 1) ? dz : sample_dist;
        float a = 1.f - expf(-delta * sigma[(long)ray * SC + i]);
        w[i] = a * T;
        T *= (1.f - a + 1e-15f);
    }
    // pdf over w[1..SC-2] (126 vals); cdf length SC-1 = 127 with cdf[0]=0
    float cdf[SC - 1];
    float sum = 0.f;
    for (int k = 1; k <= SC - 2; k++) sum += w[k] + 1e-5f;
    cdf[0] = 0.f;
    float acc = 0.f;
    for (int k = 1; k <= SC - 2; k++) { acc += (w[k] + 1e-5f) / sum; cdf[k] = acc; }

    const float ustep = (1.f - 1.f / (float)SU) / (float)(SU - 1);
    int idx = 1;
    for (int j = 0; j < SU; j++) {
        float u = 0.5f / (float)SU + ustep * (float)j;
        while (idx < SC - 1 && cdf[idx] <= u) idx++;       // searchsorted 'right'
        int below = idx - 1;
        int above = (idx < SC - 1) ? idx : SC - 2;
        float cb = cdf[below], ca = cdf[above];
        float denom = ca - cb;
        if (denom < 1e-5f) denom = 1.f;
        float t = (u - cb) / denom;
        float bb = ZNEAR + dz * (float)below + 0.5f * dz;  // analytic z_mid bins
        float ba = ZNEAR + dz * (float)above + 0.5f * dz;
        newz[(long)ray * SU + j] = bb + t * (ba - bb);
    }
}

// ---------------------------------------------------------------------------
// Per-ray: two-pointer merge of sorted coarse/fine z lists + final compositing.
// ---------------------------------------------------------------------------
__global__ __launch_bounds__(256)
void merge_kernel(const float* __restrict__ sigmaC, const float* __restrict__ sigmaF,
                  const float* __restrict__ newz,
                  float* __restrict__ zAll, float* __restrict__ wAll,
                  int* __restrict__ srcb)
{
    int ray = blockIdx.x * blockDim.x + threadIdx.x;
    if (ray >= NRAYS) return;
    const float dz = (ZFAR - ZNEAR) / (float)(SC - 1);
    const float sample_dist = (ZFAR - ZNEAR) / (float)SC;

    float zl[ST]; int sl[ST];
    const float* zf = newz + (long)ray * SU;
    int i = 0, j = 0;
    for (int k = 0; k < ST; k++) {
        float zc = ZNEAR + dz * (float)i;
        bool takeC;
        if (i >= SC) takeC = false;
        else if (j >= SU) takeC = true;
        else takeC = (zc <= zf[j]);                   // ties: coarse first (stable)
        if (takeC) { zl[k] = zc;     sl[k] = i;       i++; }
        else       { zl[k] = zf[j];  sl[k] = SC + j;  j++; }
    }
    float T = 1.f;
    for (int k = 0; k < ST; k++) {
        float delta = (k < ST - 1) ? (zl[k + 1] - zl[k]) : sample_dist;
        int s = sl[k];
        float sg = (s < SC) ? sigmaC[(long)ray * SC + s]
                            : sigmaF[(long)ray * SU + (s - SC)];
        float a = 1.f - expf(-delta * sg);
        float w = a * T;
        T *= (1.f - a + 1e-15f);
        long g = (long)ray * ST + k;
        zAll[g] = zl[k]; wAll[g] = w; srcb[g] = s;
    }
}

// ---------------------------------------------------------------------------
// Color MLP: [dir(3), geo(15)] (K padded 18->32) -> relu(64) -> sigmoid(2).
// ---------------------------------------------------------------------------
__global__ __launch_bounds__(128)
void color_mlp(const float* __restrict__ rays_d,
               const _Float16* __restrict__ geoC, const _Float16* __restrict__ geoF,
               const int* __restrict__ srcb,
               const float* __restrict__ Wc1, const float* __restrict__ bc1,
               const float* __restrict__ Wc2, const float* __restrict__ bc2,
               float* __restrict__ rgb_out, int n_tiles)
{
    __shared__ _Float16 lds_h[4][16 * HID];
    const int lane = threadIdx.x & 31;
    const int wid  = threadIdx.x >> 5;
    const int col  = lane & 15;
    const int hi   = (lane >= 16) ? 1 : 0;

    __builtin_prefetch(Wc1, 0, 0);
    __builtin_prefetch(Wc2, 0, 0);

    v16h b1[4];
#pragma unroll
    for (int t = 0; t < 4; t++) {
        v16h b = {};
#pragma unroll
        for (int e = 0; e < 16; e++) {
            int K = hi * 8 + (e < 8 ? e : e + 8);
            float v = (K < 3 + GEO) ? Wc1[K * HID + 16 * t + col] : 0.f;
            b[e] = (_Float16)v;
        }
        b1[t] = b;
    }
    v16h b2[2];
#pragma unroll
    for (int c = 0; c < 2; c++) {
        v16h b = {};
#pragma unroll
        for (int e = 0; e < 16; e++) {
            int K = 32 * c + hi * 8 + (e < 8 ? e : e + 8);
            float v = (col < 2) ? Wc2[K * 2 + col] : 0.f;
            b[e] = (_Float16)v;
        }
        b2[c] = b;
    }
    float bias1[4];
#pragma unroll
    for (int t = 0; t < 4; t++) bias1[t] = bc1[16 * t + col];
    const float bias2 = (col < 2) ? bc2[col] : 0.f;

    int wglobal = blockIdx.x * 4 + wid;
    int wstride = gridDim.x * 4;
    for (int tile = wglobal; tile < n_tiles; tile += wstride) {
        const int row = col;
        long g = (long)tile * 16 + row;
        int ray = (int)(g >> 8);
        int srcI = srcb[g];
        const _Float16* gp = (srcI < SC)
            ? (geoC + ((long)ray * SC + srcI) * GEO)
            : (geoF + ((long)ray * SU + (srcI - SC)) * GEO);

        v16h a1 = {};
        if (!hi) {
            a1[0] = (_Float16)rays_d[ray * 3 + 0];
            a1[1] = (_Float16)rays_d[ray * 3 + 1];
            a1[2] = (_Float16)rays_d[ray * 3 + 2];
#pragma unroll
            for (int e = 3; e < 8; e++) a1[e] = gp[e - 3];   // K=3..7  -> geo0..4
            a1[8] = gp[13];                                   // K=16    -> geo13
            a1[9] = gp[14];                                   // K=17    -> geo14
        } else {
#pragma unroll
            for (int e = 0; e < 8; e++) a1[e] = gp[5 + e];   // K=8..15 -> geo5..12
        }

        v8f cacc[4] = {{}, {}, {}, {}};
        cacc[0] = __builtin_amdgcn_wmma_f32_16x16x32_f16(false, a1, false, b1[0], (short)0, cacc[0], false, false);
        cacc[1] = __builtin_amdgcn_wmma_f32_16x16x32_f16(false, a1, false, b1[1], (short)0, cacc[1], false, false);
        cacc[2] = __builtin_amdgcn_wmma_f32_16x16x32_f16(false, a1, false, b1[2], (short)0, cacc[2], false, false);
        cacc[3] = __builtin_amdgcn_wmma_f32_16x16x32_f16(false, a1, false, b1[3], (short)0, cacc[3], false, false);

        _Float16* L = lds_h[wid];
#pragma unroll
        for (int t = 0; t < 4; t++)
#pragma unroll
            for (int r = 0; r < 8; r++) {
                float v = cacc[t][r] + bias1[t];
                v = v > 0.f ? v : 0.f;
                int rowm = r + hi * 8;
                L[rowm * HID + 16 * t + col] = (_Float16)v;
            }
        asm volatile("s_wait_dscnt 0" ::: "memory");

        v16h a2[2];
#pragma unroll
        for (int c = 0; c < 2; c++) {
            v16h a = {};
            int kb = 32 * c + hi * 8;
#pragma unroll
            for (int e = 0; e < 16; e++) {
                int K = kb + (e < 8 ? e : e + 8);
                a[e] = L[row * HID + K];
            }
            a2[c] = a;
        }
        v8f o = {};
        o = __builtin_amdgcn_wmma_f32_16x16x32_f16(false, a2[0], false, b2[0], (short)0, o, false, false);
        o = __builtin_amdgcn_wmma_f32_16x16x32_f16(false, a2[1], false, b2[1], (short)0, o, false, false);

#pragma unroll
        for (int r = 0; r < 8; r++) {
            int rowm = r + hi * 8;
            long g2 = (long)tile * 16 + rowm;
            if (col < 2) {
                float v = o[r] + bias2;
                rgb_out[g2 * 2 + col] = 1.f / (1.f + expf(-v));
            }
        }
    }
}

// ---------------------------------------------------------------------------
// Per-ray final reduction: image(2), depth, weights_sum  -> out[ray*4 + 0..3]
// ---------------------------------------------------------------------------
__global__ __launch_bounds__(256)
void reduce_kernel(const float* __restrict__ zAll, const float* __restrict__ wAll,
                   const float* __restrict__ rgb, float* __restrict__ out)
{
    int ray = blockIdx.x * blockDim.x + threadIdx.x;
    if (ray >= NRAYS) return;
    float im0 = 0.f, im1 = 0.f, depth = 0.f, wsum = 0.f;
    for (int k = 0; k < ST; k++) {
        long g = (long)ray * ST + k;
        float w = wAll[g];
        if (w > 1e-4f) {                         // rgbs masked where weights <= 1e-4
            im0 += w * rgb[g * 2 + 0];
            im1 += w * rgb[g * 2 + 1];
        }
        depth += w * zAll[g];
        wsum  += w;
    }
    out[ray * 4 + 0] = im0;
    out[ray * 4 + 1] = im1;
    out[ray * 4 + 2] = depth;
    out[ray * 4 + 3] = wsum;
}

// ---------------------------------------------------------------------------
extern "C" void kernel_launch(void* const* d_in, const int* in_sizes, int n_in,
                              void* d_out, int out_size, void* d_ws, size_t ws_size,
                              hipStream_t stream)
{
    (void)in_sizes; (void)n_in; (void)out_size; (void)ws_size;
    const float* rays_o = (const float*)d_in[0];
    const float* rays_d = (const float*)d_in[1];
    const float* Wd1 = (const float*)d_in[2];
    const float* bd1 = (const float*)d_in[3];
    const float* Wd2 = (const float*)d_in[4];
    const float* bd2 = (const float*)d_in[5];
    const float* Wc1 = (const float*)d_in[6];
    const float* bc1 = (const float*)d_in[7];
    const float* Wc2 = (const float*)d_in[8];
    const float* bc2 = (const float*)d_in[9];
    float* out = (float*)d_out;

    // workspace carve-up (256B aligned)
    char* ws = (char*)d_ws;
    size_t off = 0;
    auto carve = [&](size_t bytes) -> void* {
        void* p = ws + off;
        off = (off + bytes + 255) & ~(size_t)255;
        return p;
    };
    float*    sigmaC = (float*)   carve((size_t)NRAYS * SC * sizeof(float));
    _Float16* geoC   = (_Float16*)carve((size_t)NRAYS * SC * GEO * sizeof(_Float16));
    float*    newz   = (float*)   carve((size_t)NRAYS * SU * sizeof(float));
    float*    sigmaF = (float*)   carve((size_t)NRAYS * SU * sizeof(float));
    _Float16* geoF   = (_Float16*)carve((size_t)NRAYS * SU * GEO * sizeof(_Float16));
    float*    zAll   = (float*)   carve((size_t)NRAYS * ST * sizeof(float));
    float*    wAll   = (float*)   carve((size_t)NRAYS * ST * sizeof(float));
    int*      srcb   = (int*)     carve((size_t)NRAYS * ST * sizeof(int));
    float*    rgb    = (float*)   carve((size_t)NRAYS * ST * 2 * sizeof(float));

    const int tilesC = NRAYS * SC / 16;   // 65536
    const int tilesT = NRAYS * ST / 16;   // 131072

    // 1. coarse density (WMMA f32 K4 + f16 K32, TDM-staged Wd2)
    density_mlp<<<1024, 128, 0, stream>>>(rays_o, rays_d, nullptr,
                                          Wd1, bd1, Wd2, bd2,
                                          sigmaC, geoC, SC, tilesC);
    // 2. coarse composite + inverse-CDF sampling
    pdf_kernel<<<NRAYS / 256, 256, 0, stream>>>(sigmaC, newz);
    // 3. fine density (WMMA)
    density_mlp<<<1024, 128, 0, stream>>>(rays_o, rays_d, newz,
                                          Wd1, bd1, Wd2, bd2,
                                          sigmaF, geoF, SU, tilesC);
    // 4. sorted merge + final weights
    merge_kernel<<<NRAYS / 256, 256, 0, stream>>>(sigmaC, sigmaF, newz,
                                                  zAll, wAll, srcb);
    // 5. color MLP over merged samples (WMMA)
    color_mlp<<<1024, 128, 0, stream>>>(rays_d, geoC, geoF, srcb,
                                        Wc1, bc1, Wc2, bc2, rgb, tilesT);
    // 6. per-ray reduction -> output (N, 4)
    reduce_kernel<<<NRAYS / 256, 256, 0, stream>>>(zAll, wAll, rgb, out);
}
